// PhiMoE_7516192768997
// MI455X (gfx1250) — compile-verified, matrix operand order checked by reading, not run on hardware
//
#include <hip/hip_runtime.h>

#define T_TOK 1024
#define H_DIM 2048
#define I_DIM 4096
#define E_EXP 8
#define JEPS  0.01f

typedef __attribute__((ext_vector_type(16))) _Float16 v16h;
typedef __attribute__((ext_vector_type(8)))  _Float16 v8h;
typedef __attribute__((ext_vector_type(8)))  float    v8f;
typedef __attribute__((ext_vector_type(4)))  float    v4f;

#if defined(__has_builtin)
#if __has_builtin(__builtin_amdgcn_tensor_load_to_lds) && __has_builtin(__builtin_amdgcn_s_wait_tensorcnt)
#define HAVE_TDM 1
#endif
#endif
#ifndef HAVE_TDM
#define HAVE_TDM 0
#endif

#if HAVE_TDM
typedef __attribute__((ext_vector_type(4))) unsigned int u32x4;
typedef __attribute__((ext_vector_type(8))) int          i32x8;
typedef __attribute__((ext_vector_type(4))) int          i32x4;

// Issue one TDM 2D tile load: `rows` rows x 32 fp32, row pitch `stride` elems,
// into LDS at byte offset lds_off with +4dw padding per 32dw row (bank spread).
// D# layout per CDNA5 ISA ch.8 (group0/group1; groups 2-3 zero for 2D tiles).
// 6-arg builtin form (clang-23/therock): extra int32x8 (unused VADDR4) + cpol.
__device__ __forceinline__ void tdm_load_tile(unsigned lds_off, const float* gptr,
                                              int rows, int stride) {
    unsigned long long ga = (unsigned long long)(uintptr_t)gptr;
    u32x4 g0;
    g0[0] = 1u;                                            // count=1, user mode
    g0[1] = lds_off;                                       // lds_addr [63:32]
    g0[2] = (unsigned)(ga & 0xffffffffull);                // global_addr lo
    g0[3] = (unsigned)((ga >> 32) & 0x1ffffffull) | (2u << 30); // addr hi | type=2
    i32x8 g1;
    g1[0] = (2 << 16) | (1 << 20) | (4 << 22) | (3 << 25); // 4B elems, pad_en, 32dw interval, 4dw pad
    g1[1] = (32 << 16);                                    // tensor_dim0 = 32  [79:48]
    g1[2] = (rows << 16);                                  // tensor_dim1 lo16  [95:80]
    g1[3] = (32 << 16);                                    // tile_dim0 = 32    [127:112]
    g1[4] = rows;                                          // tile_dim1         [143:128]
    g1[5] = stride;                                        // tensor_dim0_stride lo32
    g1[6] = 0;
    g1[7] = 0;
    i32x4 z4 = {0, 0, 0, 0};
    i32x8 z8 = {0, 0, 0, 0, 0, 0, 0, 0};
    __builtin_amdgcn_tensor_load_to_lds(g0, g1, z4, z4, z8, 0);
}
#endif

// ---- WMMA fragment helpers (ISA 7.12.2, wave32) ----------------------------
// A (16x32 f16): lane l -> row m=l%16, K-runs [8*(l/16)..+7] and [16+8*(l/16)..+7]
__device__ __forceinline__ v16h load_h_frag(const _Float16* rowptr, int k0, int lane) {
    int half = lane >> 4;
    const _Float16* base = rowptr + k0 + 8 * half;
    v8h lo = *(const v8h*)(base);
    v8h hi = *(const v8h*)(base + 16);
    v16h a;
#pragma unroll
    for (int i = 0; i < 8; ++i) { a[i] = lo[i]; a[i + 8] = hi[i]; }
    return a;
}

__device__ __forceinline__ v16h cvt_b_frag(v4f f0, v4f f1, v4f f2, v4f f3) {
    v16h b;
#pragma unroll
    for (int j = 0; j < 4; ++j) {
        b[j]      = (_Float16)f0[j];
        b[j + 4]  = (_Float16)f1[j];
        b[j + 8]  = (_Float16)f2[j];
        b[j + 12] = (_Float16)f3[j];
    }
    return b;
}

// B frag from padded LDS tile: local row lr, row pitch 36 floats
__device__ __forceinline__ v16h load_b_frag_lds(const float* tile, int lr, int lane) {
    int half = lane >> 4;
    const float* bp = tile + lr * 36 + 8 * half;
    return cvt_b_frag(*(const v4f*)(bp), *(const v4f*)(bp + 4),
                      *(const v4f*)(bp + 16), *(const v4f*)(bp + 20));
}

// B frag straight from a global fp32 row-major matrix (fallback path)
__device__ __forceinline__ v16h load_b_frag_f32(const float* W, size_t row, size_t ld,
                                                int k0, int lane) {
    int half = lane >> 4;
    const float* base = W + row * ld + (size_t)k0 + 8 * half;
    return cvt_b_frag(*(const v4f*)(base), *(const v4f*)(base + 4),
                      *(const v4f*)(base + 16), *(const v4f*)(base + 20));
}

// ---- init: zero output + per-expert counters -------------------------------
__global__ void k_init(float* out, int* counts, int n) {
    int i = blockIdx.x * blockDim.x + threadIdx.x;
    if (i < n) out[i] = 0.f;
    if (i < E_EXP) counts[i] = 0;
}

// ---- convert hidden_states to f16 once (4 MB) ------------------------------
__global__ void k_cvt_x(const float* __restrict__ x, _Float16* __restrict__ xb, int n) {
    int i = blockIdx.x * blockDim.x + threadIdx.x;
    if (i < n) xb[i] = (_Float16)x[i];
}

// ---- router + sparsemixer + expert token-list scatter ----------------------
__global__ __launch_bounds__(256)
void k_router(const float* __restrict__ x, const float* __restrict__ gw,
              int* __restrict__ counts, int* __restrict__ lists,
              float* __restrict__ mult) {
    int t = blockIdx.x;
    int lane = threadIdx.x & 31;
    int w = threadIdx.x >> 5;
    __shared__ float logits[E_EXP];

    const float* xr = x + (size_t)t * H_DIM;
    const float* gr = gw + (size_t)w * H_DIM;
    float sum = 0.f;
    for (int k = lane; k < H_DIM; k += 32) sum += xr[k] * gr[k];
#pragma unroll
    for (int off = 16; off > 0; off >>= 1) sum += __shfl_xor(sum, off, 32);
    if (lane == 0) logits[w] = sum;
    __syncthreads();

    if (threadIdx.x == 0) {
        const float NEG_INF = -__builtin_inff();
        float s[E_EXP];
        for (int e = 0; e < E_EXP; ++e) s[e] = logits[e];

        int i1 = 0; float m1 = s[0];
        for (int e = 1; e < E_EXP; ++e) if (s[e] > m1) { m1 = s[e]; i1 = e; }
        bool keep1[E_EXP];
        float mx1 = NEG_INF;
        for (int e = 0; e < E_EXP; ++e) {
            float factor = fmaxf(fabsf(s[e]), m1);
            keep1[e] = !((m1 - s[e]) / factor > 2.f * JEPS);
            if (keep1[e] && s[e] > mx1) mx1 = s[e];
        }
        float den1 = 0.f;
        for (int e = 0; e < E_EXP; ++e) if (keep1[e]) den1 += __expf(s[e] - mx1);
        float mult1 = __expf(s[i1] - mx1) / den1;

        int i2 = 0; float m2 = NEG_INF;
        for (int e = 0; e < E_EXP; ++e) if (e != i1 && s[e] > m2) { m2 = s[e]; i2 = e; }
        bool keep2[E_EXP];
        float mx2 = NEG_INF;
        for (int e = 0; e < E_EXP; ++e) {
            float factor2 = fmaxf(fabsf(s[e]), m2);
            keep2[e] = !(((m2 - s[e]) / factor2 > 2.f * JEPS) || (e == i1));
            if (keep2[e] && s[e] > mx2) mx2 = s[e];
        }
        float den2 = 0.f;
        for (int e = 0; e < E_EXP; ++e) if (keep2[e]) den2 += __expf(s[e] - mx2);
        float mult2 = __expf(s[i2] - mx2) / den2;

        int p1 = atomicAdd(&counts[i1], 1);
        lists[i1 * T_TOK + p1] = (t << 1);
        mult[2 * t] = mult1;
        int p2 = atomicAdd(&counts[i2], 1);
        lists[i2 * T_TOK + p2] = (t << 1) | 1;
        mult[2 * t + 1] = mult2;
    }
}

// ---- pass 1: act = silu(x·Wg) * (x·Wu), gathered per expert ----------------
// block: 16 gathered tokens x 64 I-cols; waves 0-3 gate cols, waves 4-7 up cols.
// Weight tiles (64 rows x 32 k, fp32) double-buffered in LDS via TDM.
#define NK1 (H_DIM / 32)
__global__ __launch_bounds__(256)
void k_gateup(const _Float16* __restrict__ xb, const float* __restrict__ ws,
              const int* __restrict__ counts, const int* __restrict__ lists,
              _Float16* __restrict__ act) {
    int e = blockIdx.z;
    int cnt = counts[e];
    int m0 = blockIdx.y * 16;
    if (m0 >= cnt) return;
    int i0 = blockIdx.x * 64;
    int lane = threadIdx.x & 31;
    int w = threadIdx.x >> 5;
    bool is_up = (w >= 4);
    int cb = (is_up ? w - 4 : w) * 16;     // column base within the 64-col tile
    int half = lane >> 4;
    int n = lane & 15;

    __shared__ int s_row[16];
    __shared__ int s_tok[16];
#if HAVE_TDM
    __shared__ float bw[2][2][64 * 36];    // [buf][gate|up][row*36+k], padded rows
#endif
    __shared__ float uex[16][68];          // up-value exchange for silu combine

    if (threadIdx.x < 16) {
        int m = m0 + threadIdx.x;
        int entry = (m < cnt) ? lists[e * T_TOK + m] : -1;
        s_row[threadIdx.x] = entry;
        s_tok[threadIdx.x] = (entry >= 0) ? (entry >> 1) : 0;
    }
    __syncthreads();

    const float* we = ws + (size_t)e * 2 * I_DIM * H_DIM;
    const _Float16* xrow = xb + (size_t)s_tok[n] * H_DIM;
    v8f c = {};

#if HAVE_TDM
    if (w == 0) {
        tdm_load_tile((unsigned)(uintptr_t)&bw[0][0][0], we + (size_t)i0 * H_DIM, 64, H_DIM);
        tdm_load_tile((unsigned)(uintptr_t)&bw[0][1][0], we + (size_t)(I_DIM + i0) * H_DIM, 64, H_DIM);
    }
    for (int it = 0; it < NK1; ++it) {
        int k0 = it * 32;
        int cur = it & 1;
        if (w == 0) {
            if (it + 1 < NK1) {
                int k1 = k0 + 32;
                tdm_load_tile((unsigned)(uintptr_t)&bw[1 - cur][0][0],
                              we + (size_t)i0 * H_DIM + k1, 64, H_DIM);
                tdm_load_tile((unsigned)(uintptr_t)&bw[1 - cur][1][0],
                              we + (size_t)(I_DIM + i0) * H_DIM + k1, 64, H_DIM);
                __builtin_amdgcn_s_wait_tensorcnt(2);  // current buffer done, next in flight
            } else {
                __builtin_amdgcn_s_wait_tensorcnt(0);
            }
        }
        __syncthreads();
        v16h a = load_h_frag(xrow, k0, lane);
        v16h b = load_b_frag_lds(&bw[cur][is_up ? 1 : 0][0], cb + n, lane);
        c = __builtin_amdgcn_wmma_f32_16x16x32_f16(false, a, false, b, (short)0, c, false, false);
        __syncthreads();                   // buffer free before wave0 refills it
    }
#else
    size_t grow = (size_t)(i0 + cb + n) + (is_up ? (size_t)I_DIM : 0);
    for (int k0 = 0; k0 < H_DIM; k0 += 32) {
        if (k0 + 64 <= H_DIM)
            __builtin_prefetch(we + grow * H_DIM + k0 + 32, 0, 1);
        v16h a = load_h_frag(xrow, k0, lane);
        v16h b = load_b_frag_f32(we, grow, H_DIM, k0, lane);
        c = __builtin_amdgcn_wmma_f32_16x16x32_f16(false, a, false, b, (short)0, c, false, false);
    }
#endif

    // silu(g)*u combine: up waves publish, gate waves finish + store act (f16)
    if (is_up) {
#pragma unroll
        for (int r = 0; r < 8; ++r) uex[r + 8 * half][cb + n] = c[r];
    }
    __syncthreads();
    if (!is_up) {
#pragma unroll
        for (int r = 0; r < 8; ++r) {
            int m = r + 8 * half;
            int entry = s_row[m];
            if (entry >= 0) {
                float g = c[r];
                float u = uex[m][cb + n];
                float sig = __builtin_amdgcn_rcpf(1.f + __expf(-g));  // v_rcp_f32
                act[(size_t)entry * I_DIM + i0 + cb + n] = (_Float16)(g * sig * u);
            }
        }
    }
}

// ---- pass 2: out[t] += mult * (act · w2s^T), gathered per expert -----------
// block: 16 tokens x 128 h-cols; one 128x32 fp32 weight tile double-buffered via TDM.
#define NK2 (I_DIM / 32)
__global__ __launch_bounds__(256)
void k_down(const _Float16* __restrict__ act, const float* __restrict__ w2s,
            const int* __restrict__ counts, const int* __restrict__ lists,
            const float* __restrict__ mult, float* __restrict__ out) {
    int e = blockIdx.z;
    int cnt = counts[e];
    int m0 = blockIdx.y * 16;
    if (m0 >= cnt) return;
    int h0 = blockIdx.x * 128;
    int lane = threadIdx.x & 31;
    int w = threadIdx.x >> 5;
    int half = lane >> 4;
    int n = lane & 15;

    __shared__ int   s_row[16];
    __shared__ int   s_tok[16];
    __shared__ float s_mul[16];
#if HAVE_TDM
    __shared__ float bw[2][128 * 36];
#endif
    if (threadIdx.x < 16) {
        int m = m0 + threadIdx.x;
        int entry = (m < cnt) ? lists[e * T_TOK + m] : -1;
        s_row[threadIdx.x] = entry;
        s_tok[threadIdx.x] = (entry >= 0) ? (entry >> 1) : 0;
        s_mul[threadIdx.x] = (entry >= 0) ? mult[entry] : 0.f;
    }
    __syncthreads();

    const float* wd = w2s + (size_t)e * H_DIM * I_DIM;
    int ar = s_row[n];
    const _Float16* arow = act + (size_t)((ar >= 0) ? ar : 0) * I_DIM;
    v8f c = {};

#if HAVE_TDM
    if (w == 0)
        tdm_load_tile((unsigned)(uintptr_t)&bw[0][0], wd + (size_t)h0 * I_DIM, 128, I_DIM);
    for (int it = 0; it < NK2; ++it) {
        int k0 = it * 32;
        int cur = it & 1;
        if (w == 0) {
            if (it + 1 < NK2) {
                tdm_load_tile((unsigned)(uintptr_t)&bw[1 - cur][0],
                              wd + (size_t)h0 * I_DIM + k0 + 32, 128, I_DIM);
                __builtin_amdgcn_s_wait_tensorcnt(1);
            } else {
                __builtin_amdgcn_s_wait_tensorcnt(0);
            }
        }
        __syncthreads();
        v16h a = load_h_frag(arow, k0, lane);
        v16h b = load_b_frag_lds(&bw[cur][0], w * 16 + n, lane);
        c = __builtin_amdgcn_wmma_f32_16x16x32_f16(false, a, false, b, (short)0, c, false, false);
        __syncthreads();
    }
#else
    size_t rowh = (size_t)(h0 + w * 16 + n);
    for (int k0 = 0; k0 < I_DIM; k0 += 32) {
        if (k0 + 64 <= I_DIM)
            __builtin_prefetch(wd + rowh * I_DIM + k0 + 32, 0, 1);
        v16h a = load_h_frag(arow, k0, lane);
        v16h b = load_b_frag_f32(wd, rowh, I_DIM, k0, lane);
        c = __builtin_amdgcn_wmma_f32_16x16x32_f16(false, a, false, b, (short)0, c, false, false);
    }
#endif

#pragma unroll
    for (int r = 0; r < 8; ++r) {
        int m = r + 8 * half;
        if (s_row[m] >= 0) {
            float v = s_mul[m] * c[r];
            atomicAdd(&out[(size_t)s_tok[m] * H_DIM + h0 + w * 16 + n], v);
        }
    }
}

// ---- host-side launch ------------------------------------------------------
extern "C" void kernel_launch(void* const* d_in, const int* in_sizes, int n_in,
                              void* d_out, int out_size, void* d_ws, size_t ws_size,
                              hipStream_t stream) {
    const float* x   = (const float*)d_in[0];   // [T,H]
    const float* gw  = (const float*)d_in[1];   // [E,H]
    const float* ws  = (const float*)d_in[2];   // [E,2I,H]
    const float* w2s = (const float*)d_in[3];   // [E,H,I]
    float* out = (float*)d_out;                 // [T,H]

    char* p = (char*)d_ws;
    int*   counts = (int*)p;                                     // 8 ints
    float* mult   = (float*)(p + 256);                           // 2T floats
    int*   lists  = (int*)(p + 256 + 2 * T_TOK * sizeof(float)); // E*T ints
    size_t off = 256 + 2 * T_TOK * 4 + (size_t)E_EXP * T_TOK * 4;
    off = (off + 255) & ~(size_t)255;
    _Float16* xb = (_Float16*)(p + off);                         // T*H f16 (4 MB)
    off += (size_t)T_TOK * H_DIM * sizeof(_Float16);
    off = (off + 255) & ~(size_t)255;
    _Float16* act = (_Float16*)(p + off);                        // 2T*I f16 (16 MB)

    int n_out = T_TOK * H_DIM;
    k_init <<<(n_out + 255) / 256, 256, 0, stream>>>(out, counts, n_out);
    k_cvt_x<<<(T_TOK * H_DIM + 255) / 256, 256, 0, stream>>>(x, xb, T_TOK * H_DIM);
    k_router<<<T_TOK, 256, 0, stream>>>(x, gw, counts, lists, mult);

    dim3 g1(I_DIM / 64, T_TOK / 16, E_EXP);
    k_gateup<<<g1, 256, 0, stream>>>(xb, ws, counts, lists, act);

    dim3 g2(H_DIM / 128, T_TOK / 16, E_EXP);
    k_down<<<g2, 256, 0, stream>>>(act, w2s, counts, lists, mult, out);
}
